// EdgeAwareAttention_37890201485507
// MI455X (gfx1250) — compile-verified, matrix operand order checked by reading, not run on hardware
//
#include <hip/hip_runtime.h>
#include <hip/hip_bf16.h>
#include <stdint.h>

// ---------------------------------------------------------------------------
// EdgeAwareAttention for MI455X (gfx1250, wave32, WMMA, async global->LDS)
// B=8, N=512, D=256, E=16, H=8, HD=32
// ---------------------------------------------------------------------------

typedef __attribute__((ext_vector_type(16))) _Float16 v16h;
typedef __attribute__((ext_vector_type(8)))  _Float16 v8h;
typedef __attribute__((ext_vector_type(8)))  float    v8f;
typedef __attribute__((ext_vector_type(4)))  float    v4f;

#define BB  8
#define NN  512
#define DD  256
#define EE  16
#define HH  8
#define HDD 32

// ---- WMMA fragment loaders ------------------------------------------------
// 16-bit A/B fragment layout (ISA 7.12.2):
//   lane l: row/col = l&15 ;  elements 0..7  -> K = (l>>4)*8 .. +7
//                             elements 8..15 -> K = 16 + (l>>4)*8 .. +7

__device__ __forceinline__ v16h frag_from_f32(const float* base, int stride) {
  int lane = threadIdx.x & 31;
  int r = lane & 15, hi = lane >> 4;
  const float* p = base + r * stride + hi * 8;
  v16h f;
#pragma unroll
  for (int i = 0; i < 8; ++i) { f[i] = (_Float16)p[i]; f[8 + i] = (_Float16)p[16 + i]; }
  return f;
}

__device__ __forceinline__ v16h frag_from_h(const _Float16* base, int stride) {
  int lane = threadIdx.x & 31;
  int r = lane & 15, hi = lane >> 4;
  const _Float16* p = base + r * stride + hi * 8;
  v8h lo = *(const v8h*)p;          // K = khalf .. khalf+7   (16B aligned)
  v8h hh = *(const v8h*)(p + 16);   // K = 16+khalf .. +7
  v16h f;
#pragma unroll
  for (int i = 0; i < 8; ++i) { f[i] = lo[i]; f[8 + i] = hh[i]; }
  return f;
}

// ---- async global -> LDS copy (CDNA5, tracked by ASYNCcnt) ----------------
// Copies 128 bytes (8 x b128) per calling thread, VGPR-bypass.
// ISA 10 / 15.18: INST_OFFSET is added to BOTH the LDS and global addresses.
__device__ __forceinline__ void async_copy128B(float* lds_dst, const float* gsrc) {
  uint32_t ldsoff = (uint32_t)(uintptr_t)lds_dst;   // low 32 bits = LDS addr
  uint64_t g = (uint64_t)(uintptr_t)gsrc;
#pragma unroll
  for (int i = 0; i < 8; ++i)
    asm volatile("global_load_async_to_lds_b128 %0, %1, off offset:%c2"
                 :: "v"(ldsoff), "v"(g), "i"(i * 16) : "memory");
}

__device__ __forceinline__ void wait_async0() {
  asm volatile("s_wait_asynccnt 0x0" ::: "memory");
}

// ---------------------------------------------------------------------------
// Kernel 1: QKV projection.  y = x @ W^T + b  (contraction over D, both
// operands row-contiguous over the contracted dim -> identical frag loads).
// grid = (256 row-tiles, 16 col-tiles, 3 matrices), block = 1 wave.
// Q,K stored f16 [B,H,N,HD]; V stored f16 transposed [B,H,HD,N].
// ---------------------------------------------------------------------------
__global__ void __launch_bounds__(32) qkv_proj_kernel(
    const float* __restrict__ x,
    const float* __restrict__ Wq, const float* __restrict__ bq,
    const float* __restrict__ Wk, const float* __restrict__ bk,
    const float* __restrict__ Wv, const float* __restrict__ bv,
    _Float16* __restrict__ q_ws, _Float16* __restrict__ k_ws,
    _Float16* __restrict__ v_t)
{
  int rt = blockIdx.x;          // row tile over B*N
  int ct = blockIdx.y;          // col tile over D
  int m  = blockIdx.z;          // 0=Q 1=K 2=V
  const float* W  = (m == 0) ? Wq : (m == 1) ? Wk : Wv;
  const float* bi = (m == 0) ? bq : (m == 1) ? bk : bv;

  int lane = threadIdx.x;
  int r = lane & 15, hi = lane >> 4;

  const float* xbase = x + (size_t)rt * 16 * DD;
  const float* wbase = W + (size_t)ct * 16 * DD;

  v8f acc = {};
#pragma unroll
  for (int kt = 0; kt < DD / 32; ++kt) {
    v16h a = frag_from_f32(xbase + kt * 32, DD);
    v16h b = frag_from_f32(wbase + kt * 32, DD);
    acc = __builtin_amdgcn_wmma_f32_16x16x32_f16(false, a, false, b,
                                                 (short)0, acc, false, false);
  }

  int o  = ct * 16 + r;               // output column in D
  float bv_o = bi[o];
  int h = o >> 5, d = o & 31;
#pragma unroll
  for (int j = 0; j < 8; ++j) {       // C layout: (M=j+8*hi, N=r)
    int g  = rt * 16 + j + 8 * hi;    // global row in B*N
    int bb = g >> 9, n = g & (NN - 1);
    _Float16 val = (_Float16)(acc[j] + bv_o);
    if (m == 2) v_t[((size_t)(bb * HH + h) * HDD + d) * NN + n] = val;
    else {
      _Float16* dst = (m == 0) ? q_ws : k_ws;
      dst[((size_t)(bb * HH + h) * NN + n) * HDD + d] = val;
    }
  }
}

// ---------------------------------------------------------------------------
// Kernel 2: fused edge-bias + flash attention.
// block = 256 threads = 8 waves; wave w = head w; block handles one
// (batch b, q-tile of 16).  edge_attr slice is streamed to LDS ONCE per
// block with double-buffered ASYNC copies (overlaps HBM with WMMA/softmax),
// shared by all 8 heads.
// ---------------------------------------------------------------------------
__global__ void __launch_bounds__(256) edge_attn_kernel(
    const _Float16* __restrict__ q_ws, const _Float16* __restrict__ k_ws,
    const _Float16* __restrict__ v_t,  const float* __restrict__ edge,
    const float* __restrict__ We,      const float* __restrict__ be,
    const unsigned char* __restrict__ mask, _Float16* __restrict__ o_ws)
{
  __shared__ __align__(16) float    elds[2][16 * 32 * EE];  // 2 x 32 KB: [q][k][e]
  __shared__ __align__(16) _Float16 pbuf[HH][16 * 32];      // 8 KB: per-wave P

  int qt  = blockIdx.x;          // q tile (0..31)
  int b   = blockIdx.y;          // batch
  int tid = threadIdx.x;
  int w   = tid >> 5;            // wave id == head
  int lane = tid & 31;
  int r = lane & 15, hi = lane >> 4;
  const int h = w;

  float we[EE];
#pragma unroll
  for (int e = 0; e < EE; ++e) we[e] = We[e * HH + h];
  const float beh = be[h];

  const _Float16* qb = q_ws + ((size_t)(b * HH + h) * NN + qt * 16) * HDD;
  const _Float16* kb = k_ws + (size_t)(b * HH + h) * NN * HDD;
  const _Float16* vb = v_t  + (size_t)(b * HH + h) * HDD * NN;

  v16h qfrag = frag_from_h(qb, HDD);   // Q tile resident for whole loop

  v8f O0 = {}, O1 = {};
  float mrow[8], lrow[8];
#pragma unroll
  for (int j = 0; j < 8; ++j) { mrow[j] = -3.0e38f; lrow[j] = 0.f; }
  const float scale = 0.17677669529663687f;   // 1/sqrt(32)

  // cooperative-copy roles: thread copies 128B of edge_attr per chunk
  const int qq = tid >> 4, seg = tid & 15;
  const float* esrc0 = edge + (((size_t)(b * NN + qt * 16 + qq)) * NN) * EE + seg * 32;

  // prologue: async-stage chunk 0 into buffer 0
  async_copy128B(&elds[0][qq * 512 + seg * 32], esrc0);

  for (int c = 0; c < NN / 32; ++c) {
    int c0 = c * 32;
    int cur = c & 1;

    wait_async0();       // my slice of chunk c is in LDS
    __syncthreads();     // everyone's slice is in LDS

    // async-stage chunk c+1 into the other buffer (overlaps with compute)
    if (c + 1 < NN / 32)
      async_copy128B(&elds[cur ^ 1][qq * 512 + seg * 32],
                     esrc0 + (size_t)(c0 + 32) * EE);

    // ---- scores: S[16q x 32k] via two WMMAs (K = HD = 32 exactly) ----
    v16h kf0 = frag_from_h(kb + (size_t)c0 * HDD, HDD);
    v16h kf1 = frag_from_h(kb + (size_t)(c0 + 16) * HDD, HDD);
    v8f S0 = {}, S1 = {};
    S0 = __builtin_amdgcn_wmma_f32_16x16x32_f16(false, qfrag, false, kf0,
                                                (short)0, S0, false, false);
    S1 = __builtin_amdgcn_wmma_f32_16x16x32_f16(false, qfrag, false, kf1,
                                                (short)0, S1, false, false);

    float mk0 = mask[b * NN + c0 + r]      ? 0.f : -3.0e38f;
    float mk1 = mask[b * NN + c0 + 16 + r] ? 0.f : -3.0e38f;

    // ---- edge bias + online softmax (per C-layout element) ----
#pragma unroll
    for (int j = 0; j < 8; ++j) {
      int qrow = j + 8 * hi;                       // M of this acc element
      const float* e0 = &elds[cur][(qrow * 32 + r) * EE];
      const float* e1 = &elds[cur][(qrow * 32 + 16 + r) * EE];
      float b0 = beh, b1 = beh;
#pragma unroll
      for (int e = 0; e < EE; ++e) { b0 += e0[e] * we[e]; b1 += e1[e] * we[e]; }
      float s0 = S0[j] * scale + b0 + mk0;
      float s1 = S1[j] * scale + b1 + mk1;

      float mx = fmaxf(s0, s1);                    // row-max across 16 lanes
#pragma unroll
      for (int d = 1; d < 16; d <<= 1) mx = fmaxf(mx, __shfl_xor(mx, d, 32));
      float mnew = fmaxf(mrow[j], mx);
      float corr = __expf(mrow[j] - mnew);
      float p0 = __expf(s0 - mnew);
      float p1 = __expf(s1 - mnew);
      float ps = p0 + p1;                          // row-sum across 16 lanes
#pragma unroll
      for (int d = 1; d < 16; d <<= 1) ps += __shfl_xor(ps, d, 32);
      lrow[j] = lrow[j] * corr + ps;
      mrow[j] = mnew;
      O0[j] *= corr;  O1[j] *= corr;
      pbuf[w][qrow * 32 + r]      = (_Float16)p0;  // C layout -> LDS
      pbuf[w][qrow * 32 + 16 + r] = (_Float16)p1;
    }
    __syncthreads();   // publish P; also fences edge reads of this chunk

    // ---- O += P x V : two WMMAs (dims 0..15, 16..31) ----
    v16h pf  = frag_from_h(&pbuf[w][0], 32);
    v16h vf0 = frag_from_h(vb + (size_t)0  * NN + c0, NN);
    v16h vf1 = frag_from_h(vb + (size_t)16 * NN + c0, NN);
    O0 = __builtin_amdgcn_wmma_f32_16x16x32_f16(false, pf, false, vf0,
                                                (short)0, O0, false, false);
    O1 = __builtin_amdgcn_wmma_f32_16x16x32_f16(false, pf, false, vf1,
                                                (short)0, O1, false, false);
  }

  // ---- normalize and store f16 [B,N,D] (D col = h*32 + d) ----
#pragma unroll
  for (int j = 0; j < 8; ++j) {
    float inv = 1.0f / fmaxf(lrow[j], 1e-20f);
    int qrow = qt * 16 + j + 8 * hi;
    size_t rowoff = ((size_t)(b * NN + qrow)) * DD + h * 32;
    o_ws[rowoff + r]      = (_Float16)(O0[j] * inv);
    o_ws[rowoff + 16 + r] = (_Float16)(O1[j] * inv);
  }
}

// ---------------------------------------------------------------------------
// Kernel 3: output projection  out = attn_out @ Wo^T + bo  (f32 result)
// ---------------------------------------------------------------------------
__global__ void __launch_bounds__(32) out_proj_kernel(
    const _Float16* __restrict__ o_ws, const float* __restrict__ Wo,
    const float* __restrict__ bo, float* __restrict__ out)
{
  int rt = blockIdx.x, ct = blockIdx.y;
  int lane = threadIdx.x;
  int r = lane & 15, hi = lane >> 4;

  const _Float16* abase = o_ws + (size_t)rt * 16 * DD;
  const float*    wbase = Wo  + (size_t)ct * 16 * DD;

  v8f acc = {};
#pragma unroll
  for (int kt = 0; kt < DD / 32; ++kt) {
    v16h a = frag_from_h(abase + kt * 32, DD);
    v16h b = frag_from_f32(wbase + kt * 32, DD);
    acc = __builtin_amdgcn_wmma_f32_16x16x32_f16(false, a, false, b,
                                                 (short)0, acc, false, false);
  }
  int o = ct * 16 + r;
  float bb = bo[o];
#pragma unroll
  for (int j = 0; j < 8; ++j) {
    int g = rt * 16 + j + 8 * hi;
    out[(size_t)g * DD + o] = acc[j] + bb;
  }
}

// ---------------------------------------------------------------------------
extern "C" void kernel_launch(void* const* d_in, const int* in_sizes, int n_in,
                              void* d_out, int out_size, void* d_ws, size_t ws_size,
                              hipStream_t stream) {
  (void)in_sizes; (void)n_in; (void)out_size; (void)ws_size;

  const float* x    = (const float*)d_in[0];
  const float* edge = (const float*)d_in[1];
  const unsigned char* mask = (const unsigned char*)d_in[2];  // jnp bool = 1B
  const float* Wq = (const float*)d_in[3];
  const float* bq = (const float*)d_in[4];
  const float* Wk = (const float*)d_in[5];
  const float* bk = (const float*)d_in[6];
  const float* Wv = (const float*)d_in[7];
  const float* bv = (const float*)d_in[8];
  const float* Wo = (const float*)d_in[9];
  const float* bo = (const float*)d_in[10];
  const float* We = (const float*)d_in[11];
  const float* be = (const float*)d_in[12];
  float* out = (float*)d_out;

  // workspace: 4 x (B*N*D) f16 buffers = 8 MB total
  const size_t SZ = (size_t)BB * NN * DD;   // 1,048,576 elements each
  _Float16* ws   = (_Float16*)d_ws;
  _Float16* q_ws = ws;
  _Float16* k_ws = ws + SZ;
  _Float16* v_t  = ws + 2 * SZ;
  _Float16* o_ws = ws + 3 * SZ;

  // 1) QKV projections (WMMA GEMM, f32 -> f16)
  qkv_proj_kernel<<<dim3((BB * NN) / 16, DD / 16, 3), 32, 0, stream>>>(
      x, Wq, bq, Wk, bk, Wv, bv, q_ws, k_ws, v_t);

  // 2) fused edge-bias + flash attention (edge_attr streamed async, once)
  edge_attn_kernel<<<dim3(NN / 16, BB), 256, 0, stream>>>(
      q_ws, k_ws, v_t, edge, We, be, mask, o_ws);

  // 3) output projection
  out_proj_kernel<<<dim3((BB * NN) / 16, DD / 16), 32, 0, stream>>>(
      o_ws, Wo, bo, out);
}
